// VectorQuantizerEMA_86887188398520
// MI455X (gfx1250) — matrix operand order repaired and checked by reading, not compiled, not run on hardware
//
#include <hip/hip_runtime.h>
#include <hip/hip_bf16.h>
#include <float.h>
#include <math.h>

typedef __attribute__((ext_vector_type(16))) __bf16 v16bf;
typedef __attribute__((ext_vector_type(8)))  float  v8f;

#define K_CODES 512
#define DIM     64
#define N_ROWS  131072      // 32*64*64
#define BATCH   32
#define HW      4096        // 64*64
#define ND      8388608     // N_ROWS*DIM

#define WMMA_BF16(A, B, C) \
    __builtin_amdgcn_wmma_f32_16x16x32_bf16(false, (A), false, (B), (short)0, (C), false, false)

// ---------------------------------------------------------------------------
// Prep: split codebook into bf16 hi/lo pair (Markidis split) + ||e||^2
// ---------------------------------------------------------------------------
__global__ void k_prep_split(const float* __restrict__ emb,
                             __bf16* __restrict__ eh, __bf16* __restrict__ el) {
    int i = blockIdx.x * blockDim.x + threadIdx.x;      // K_CODES*DIM = 32768
    float w = emb[i];
    __bf16 h = (__bf16)w;
    eh[i] = h;
    el[i] = (__bf16)(w - (float)h);
}

__global__ void k_prep_esq(const float* __restrict__ emb, float* __restrict__ esq) {
    int k = threadIdx.x;                                 // 512 threads
    const float* r = emb + k * DIM;
    float s = 0.0f;
    #pragma unroll
    for (int d = 0; d < DIM; ++d) { float w = r[d]; s += w * w; }
    esq[k] = s;
}

// ---------------------------------------------------------------------------
// B-tile register buffer for one 16-code chunk
// ---------------------------------------------------------------------------
struct BTiles { v16bf bh0, bh1, bl0, bl1; float es; };

// ---------------------------------------------------------------------------
// Nearest-code search via split-bf16 WMMA.
//   score(m,k) = ||e_k||^2 - 2 * x_m . e_k    (||x||^2 dropped: argmin-invariant)
//   A = -2x split hi/lo, B = e split hi/lo; acc = Ahi*Bhi + Alo*Bhi + Ahi*Blo
// Workgroup = 256 threads (8 waves). Each wave owns TWO 16-row A tiles (M=32):
// both tiles share every B tile (halves LDS traffic) and give two independent
// 6-deep WMMA chains (2-way intra-wave ILP x 4 waves/SIMD covers the bf16
// WMMA RAW hazard). The chunk loop is 2x unrolled with two disjoint B register
// buffers so next-chunk ds_loads issue before the current WMMA block -- LDS
// latency hides under the matrix pipe instead of s_wait_dscnt stalls.
// Full bf16 hi/lo codebook (128KB) + esq (2KB) staged in dynamic LDS (CDNA5
// WGP has 320KB LDS -> 2 workgroups of 130KB co-resident per WGP).
// ---------------------------------------------------------------------------
__global__ __launch_bounds__(256) void k_argmin(
    const float* __restrict__ x,
    const __bf16* __restrict__ g_eh, const __bf16* __restrict__ g_el,
    const float* __restrict__ g_esq, int* __restrict__ idx)
{
    extern __shared__ char smem[];
    __bf16* eh  = (__bf16*)smem;
    __bf16* el  = eh + K_CODES * DIM;
    float*  esq = (float*)(el + K_CODES * DIM);

    int tid  = threadIdx.x;
    int wv   = tid >> 5, lane = tid & 31;
    int m0   = blockIdx.x * 256 + wv * 32;        // 32 rows per wave
    int hsel = (lane < 16) ? 0 : 16;              // A-matrix 16-bit layout half
    int r0   = m0 + (lane & 15);
    int r1   = r0 + 16;
    const float* xr0 = x + (size_t)r0 * DIM;
    const float* xr1 = x + (size_t)r1 * DIM;

    // prefetch A rows (global_prefetch_b8) while we stage the codebook
    __builtin_prefetch(xr0, 0, 0);
    __builtin_prefetch(xr0 + 32, 0, 0);
    __builtin_prefetch(xr1, 0, 0);
    __builtin_prefetch(xr1 + 32, 0, 0);

    // cooperative stage of codebook: 128-bit copies
    {
        const uint4* sh = (const uint4*)g_eh;
        const uint4* sl = (const uint4*)g_el;
        uint4* dh = (uint4*)eh;
        uint4* dl = (uint4*)el;
        for (int i = tid; i < K_CODES * DIM / 8; i += 256) { dh[i] = sh[i]; dl[i] = sl[i]; }
        for (int i = tid; i < K_CODES; i += 256) esq[i] = g_esq[i];
    }
    __syncthreads();

    // Load both A tiles (=-2x) once per wave, split into bf16 hi/lo.
    // A 16x32 layout (contiguous, matches B): elems 0..15 -> cols kb+16h+0..15
    v16bf a0h[2], a0l[2], a1h[2], a1l[2];
    #pragma unroll
    for (int t = 0; t < 2; ++t) {
        const float* xr = t ? xr1 : xr0;
        v16bf* ahp = t ? a1h : a0h;
        v16bf* alp = t ? a1l : a0l;
        #pragma unroll
        for (int kb2 = 0; kb2 < 2; ++kb2) {
            int cbase = kb2 * 32 + hsel;
            #pragma unroll
            for (int half = 0; half < 2; ++half) {
                const float4* p = (const float4*)(xr + cbase + half * 8);
                float4 f0 = p[0], f1 = p[1];
                float v[8] = {f0.x, f0.y, f0.z, f0.w, f1.x, f1.y, f1.z, f1.w};
                #pragma unroll
                for (int i = 0; i < 8; ++i) {
                    float tv = -2.0f * v[i];
                    __bf16 h = (__bf16)tv;
                    ahp[kb2][half * 8 + i] = h;
                    alp[kb2][half * 8 + i] = (__bf16)(tv - (float)h);
                }
            }
        }
    }

    int nn     = lane & 15;                       // C/D layout: col = lane%16
    int bh_off = (lane < 16) ? 0 : 16;            // B 32x16: lanes>=16 hold K=16..31

    float bv0[8], bv1[8]; int bi0[8], bi1[8];
    #pragma unroll
    for (int j = 0; j < 8; ++j) { bv0[j] = FLT_MAX; bi0[j] = 0; bv1[j] = FLT_MAX; bi1[j] = 0; }

    // helper lambdas ---------------------------------------------------------
    auto load_b = [&](int code, BTiles& b) {
        b.bh0 = *(const v16bf*)(eh + code * DIM + bh_off);
        b.bh1 = *(const v16bf*)(eh + code * DIM + 32 + bh_off);
        b.bl0 = *(const v16bf*)(el + code * DIM + bh_off);
        b.bl1 = *(const v16bf*)(el + code * DIM + 32 + bh_off);
        b.es  = esq[code];
    };
    auto compute = [&](const BTiles& b, int code) {
        float es = b.es;
        v8f acc0 = {es, es, es, es, es, es, es, es};
        v8f acc1 = acc0;
        // two independent 6-deep chains, interleaved
        acc0 = WMMA_BF16(a0h[0], b.bh0, acc0);  acc1 = WMMA_BF16(a1h[0], b.bh0, acc1);
        acc0 = WMMA_BF16(a0l[0], b.bh0, acc0);  acc1 = WMMA_BF16(a1l[0], b.bh0, acc1);
        acc0 = WMMA_BF16(a0h[0], b.bl0, acc0);  acc1 = WMMA_BF16(a1h[0], b.bl0, acc1);
        acc0 = WMMA_BF16(a0h[1], b.bh1, acc0);  acc1 = WMMA_BF16(a1h[1], b.bh1, acc1);
        acc0 = WMMA_BF16(a0l[1], b.bh1, acc0);  acc1 = WMMA_BF16(a1l[1], b.bh1, acc1);
        acc0 = WMMA_BF16(a0h[1], b.bl1, acc0);  acc1 = WMMA_BF16(a1h[1], b.bl1, acc1);
        #pragma unroll
        for (int j = 0; j < 8; ++j) {
            float d0 = acc0[j];
            if (d0 < bv0[j]) { bv0[j] = d0; bi0[j] = code; }
            float d1 = acc1[j];
            if (d1 < bv1[j]) { bv1[j] = d1; bi1[j] = code; }
        }
    };
    // ------------------------------------------------------------------------

    // software-pipelined chunk loop: 32 chunks of 16 codes, unrolled x2 with
    // two B register buffers; loads for chunk c+1 / c+2 issue before the
    // WMMA blocks that consume buffers 0 / 1. Last prefetch wraps (in-bounds,
    // result unused) to keep the loop branch-free.
    BTiles b0, b1;
    load_b(nn, b0);                               // chunk 0
    for (int kc = 0; kc < 32; kc += 2) {
        load_b((kc + 1) * 16 + nn, b1);           // chunk kc+1
        compute(b0, kc * 16 + nn);                // consume buffer 0
        load_b((((kc + 2) & 31) * 16) + nn, b0);  // chunk kc+2 (wraps on last)
        compute(b1, (kc + 1) * 16 + nn);          // consume buffer 1
    }

    // min-reduce across the 16 lanes of each half (row m lives on lanes
    // (m/8)*16 .. +15 in VGPR m%8); tie-break to smaller index like argmin.
    #pragma unroll
    for (int j = 0; j < 8; ++j) {
        float v0 = bv0[j]; int i0 = bi0[j];
        float v1 = bv1[j]; int i1 = bi1[j];
        #pragma unroll
        for (int off = 1; off < 16; off <<= 1) {
            float ov0 = __shfl_xor(v0, off, 32);
            int   oi0 = __shfl_xor(i0, off, 32);
            if (ov0 < v0 || (ov0 == v0 && oi0 < i0)) { v0 = ov0; i0 = oi0; }
            float ov1 = __shfl_xor(v1, off, 32);
            int   oi1 = __shfl_xor(i1, off, 32);
            if (ov1 < v1 || (ov1 == v1 && oi1 < i1)) { v1 = ov1; i1 = oi1; }
        }
        bi0[j] = i0; bi1[j] = i1;
    }
    if (lane == 0) {
        #pragma unroll
        for (int j = 0; j < 8; ++j) { idx[m0 + j] = bi0[j]; idx[m0 + 16 + j] = bi1[j]; }
    }
    if (lane == 16) {
        #pragma unroll
        for (int j = 0; j < 8; ++j) { idx[m0 + 8 + j] = bi0[j]; idx[m0 + 24 + j] = bi1[j]; }
    }
}

// ---------------------------------------------------------------------------
// counts / dw accumulation (segment_sum) via global f32 atomics
// ---------------------------------------------------------------------------
__global__ void k_stats(const float* __restrict__ x, const int* __restrict__ idx,
                        float* __restrict__ dw, float* __restrict__ counts) {
    long long stride = (long long)gridDim.x * blockDim.x;
    for (long long i = (long long)blockIdx.x * blockDim.x + threadIdx.x;
         i < (long long)ND; i += stride) {
        int row = (int)(i >> 6), d = (int)(i & 63);
        int k = idx[row];
        atomicAdd(&dw[k * DIM + d], x[i]);
        if (d == 0) atomicAdd(&counts[k], 1.0f);
    }
}

// ---------------------------------------------------------------------------
// EMA update + Laplace smoothing + new codebook + perplexity (one WG, K thr)
// ---------------------------------------------------------------------------
__global__ void k_finalize(const float* __restrict__ counts,
                           const float* __restrict__ ema_cs,
                           const float* __restrict__ ema_w,
                           const float* __restrict__ dw,
                           float* __restrict__ new_emb,
                           float* __restrict__ out_perp) {
    __shared__ float red[K_CODES];
    int k = threadIdx.x;
    float cnt = counts[k];
    float cs  = ema_cs[k] * 0.99f + 0.01f * cnt;
    red[k] = cs; __syncthreads();
    for (int s = 256; s > 0; s >>= 1) { if (k < s) red[k] += red[k + s]; __syncthreads(); }
    float n = red[0]; __syncthreads();
    float csm = (cs + 1e-5f) / (n + (float)K_CODES * 1e-5f) * n;
    float inv = 1.0f / csm;
    for (int d = 0; d < DIM; ++d) {
        float nw = ema_w[k * DIM + d] * 0.99f + 0.01f * dw[k * DIM + d];
        new_emb[k * DIM + d] = nw * inv;
    }
    float p = cnt / (float)N_ROWS;
    red[k] = p * logf(p + 1e-10f);
    __syncthreads();
    for (int s = 256; s > 0; s >>= 1) { if (k < s) red[k] += red[k + s]; __syncthreads(); }
    if (k == 0) *out_perp = expf(-red[0]);
}

// ---------------------------------------------------------------------------
// Quantize + NHWC->NCHW transpose. quantized_st == quantized numerically.
// One block per (b, d); codebook column staged in LDS; coalesced NT stores
// (write-once output: keep it out of L2 so inputs/codebook stay resident).
// ---------------------------------------------------------------------------
__global__ void k_quant(const float* __restrict__ new_emb, const int* __restrict__ idx,
                        float* __restrict__ outq) {
    __shared__ float col[K_CODES];
    int b = blockIdx.x >> 6, d = blockIdx.x & 63;
    for (int i = threadIdx.x; i < K_CODES; i += 256) col[i] = new_emb[i * DIM + d];
    __syncthreads();
    const int* ir = idx + b * HW;
    float* orow = outq + ((size_t)(b * DIM + d)) * HW;
    for (int hw = threadIdx.x; hw < HW; hw += 256)
        __builtin_nontemporal_store(col[ir[hw]], &orow[hw]);
}

// ---------------------------------------------------------------------------
// Commitment-loss reduction (NHWC order -> coalesced reads of inputs)
// ---------------------------------------------------------------------------
__global__ void k_loss(const float* __restrict__ x, const int* __restrict__ idx,
                       const float* __restrict__ new_emb, float* __restrict__ accum) {
    __shared__ float wred[8];
    float s = 0.0f;
    long long stride = (long long)gridDim.x * blockDim.x;
    for (long long i = (long long)blockIdx.x * blockDim.x + threadIdx.x;
         i < (long long)ND; i += stride) {
        int row = (int)(i >> 6), d = (int)(i & 63);
        float diff = new_emb[idx[row] * DIM + d] - x[i];
        s += diff * diff;
    }
    for (int off = 16; off; off >>= 1) s += __shfl_xor(s, off, 32);
    int lane = threadIdx.x & 31, wv = threadIdx.x >> 5;
    if (lane == 0) wred[wv] = s;
    __syncthreads();
    if (wv == 0) {
        s = (lane < 8) ? wred[lane] : 0.0f;
        for (int off = 4; off; off >>= 1) s += __shfl_xor(s, off, 32);
        if (lane == 0) atomicAdd(accum, s);
    }
}

__global__ void k_loss_final(const float* __restrict__ accum, float* __restrict__ out_loss) {
    *out_loss = 0.25f * (*accum) / (float)ND;
}

// ---------------------------------------------------------------------------
// One-hot encodings writer: 268MB streaming write (bandwidth-dominant kernel).
// Pure integer path (0x3f800000 bit pattern), 16B/thread via two u64
// non-temporal stores (the encodings offset is only 8-byte aligned).
// ---------------------------------------------------------------------------
__global__ void k_onehot(const int* __restrict__ idx, unsigned long long* __restrict__ enc) {
    const unsigned long long FONE = 0x3f800000ull;
    int t   = blockIdx.x * blockDim.x + threadIdx.x;  // N*K/4 = 16777216 threads
    int row = t >> 7;                                 // 128 quads per row of K=512
    int k0  = (t & 127) * 4;
    int d   = idx[row] - k0;                          // 0..3 iff this quad holds the 1
    unsigned long long lo = 0, hi = 0;
    if (d == 0) lo = FONE;
    else if (d == 1) lo = FONE << 32;
    else if (d == 2) hi = FONE;
    else if (d == 3) hi = FONE << 32;
    __builtin_nontemporal_store(lo, &enc[2 * t]);
    __builtin_nontemporal_store(hi, &enc[2 * t + 1]);
}

// ---------------------------------------------------------------------------
extern "C" void kernel_launch(void* const* d_in, const int* in_sizes, int n_in,
                              void* d_out, int out_size, void* d_ws, size_t ws_size,
                              hipStream_t stream) {
    (void)in_sizes; (void)n_in; (void)out_size; (void)ws_size;
    const float* x      = (const float*)d_in[0];   // [32,64,64,64]
    const float* emb    = (const float*)d_in[1];   // [512,64]
    const float* ema_cs = (const float*)d_in[2];   // [512]
    const float* ema_w  = (const float*)d_in[3];   // [512,64]

    float* out      = (float*)d_out;
    float* out_loss = out;
    float* out_q    = out + 1;
    float* out_perp = out + 1 + (size_t)ND;
    float* out_enc  = out + 2 + (size_t)ND;

    // workspace layout (256B-aligned regions)
    char* ws = (char*)d_ws;
    float*  loss_acc = (float*)(ws + 0);          // 1 f32
    float*  counts   = (float*)(ws + 256);        // 512 f32
    float*  esq      = (float*)(ws + 2560);       // 512 f32
    float*  dw       = (float*)(ws + 4864);       // 512*64 f32 (ends 135936)
    float*  new_emb  = (float*)(ws + 136192);     // 512*64 f32 (ends 267264)
    __bf16* eh       = (__bf16*)(ws + 267520);    // 512*64 bf16 (ends 333056)
    __bf16* el       = (__bf16*)(ws + 333312);    // 512*64 bf16 (ends 398848)
    int*    idx      = (int*)(ws + 399104);       // 131072 i32 (ends 923392)

    hipMemsetAsync(ws, 0, 136192, stream);        // zero loss_acc, counts, dw

    k_prep_split<<<128, 256, 0, stream>>>(emb, eh, el);
    k_prep_esq  <<<1, K_CODES, 0, stream>>>(emb, esq);

    // 512 WGs * 8 waves * 32 rows = 131072 rows; 133120B dynamic LDS
    k_argmin<<<N_ROWS / 256, 256, 133120, stream>>>(x, eh, el, esq, idx);

    k_stats   <<<512, 256, 0, stream>>>(x, idx, dw, counts);
    k_finalize<<<1, K_CODES, 0, stream>>>(counts, ema_cs, ema_w, dw, new_emb, out_perp);
    k_quant   <<<BATCH * DIM, 256, 0, stream>>>(new_emb, idx, out_q);
    k_loss    <<<512, 256, 0, stream>>>(x, idx, new_emb, loss_acc);
    k_loss_final<<<1, 1, 0, stream>>>(loss_acc, out_loss);
    k_onehot  <<<65536, 256, 0, stream>>>(idx, (unsigned long long*)out_enc);
}
